// UpBlock_88914412961975
// MI455X (gfx1250) — matrix operand order repaired and checked by reading, not compiled, not run on hardware
//
#include <hip/hip_runtime.h>

// ---------------- problem constants (from reference setup_inputs) -----------
#define B_   2
#define M_   16384     // query points
#define N_   4096      // coarse points
#define C_   128       // x channels
#define CS_  256       // coarse feature channels
#define K1_  384       // C_ + CS_
#define DO_  128       // output channels

typedef __attribute__((ext_vector_type(16))) _Float16 v16h;
typedef __attribute__((ext_vector_type(8)))  float    v8f;

struct Frag32 { uint4 x, y; };           // 32 bytes = 16 f16 = one WMMA operand

// ---------------- f16 helpers (hardware cvt) --------------------------------
__device__ __forceinline__ unsigned short f2h(float f) {
    return __builtin_bit_cast(unsigned short, (_Float16)f);
}
__device__ __forceinline__ float h2f(unsigned short s) {
    return (float)__builtin_bit_cast(_Float16, s);
}
__device__ __forceinline__ float hlo(unsigned u) { return h2f((unsigned short)(u & 0xFFFFu)); }
__device__ __forceinline__ float hhi(unsigned u) { return h2f((unsigned short)(u >> 16)); }
__device__ __forceinline__ unsigned packh2(float lo, float hi) {
    return (unsigned)f2h(lo) | ((unsigned)f2h(hi) << 16);
}

// ---------------- 1) tiled transpose + f32->f16 convert ---------------------
// src: [rows, cols] f32 per batch;  dst[(n)*dstStride + dstColOff + c] = src[c*cols + n]
__global__ void transpose_cvt(const float* __restrict__ src, unsigned short* __restrict__ dst,
                              int cols, long srcBS, long dstBS, int dstStride, int dstColOff) {
    __shared__ float tile[32][33];
    const int b  = blockIdx.z;
    const int c0 = blockIdx.y * 32;
    const int n0 = blockIdx.x * 32;
    const float* s = src + (long)b * srcBS;
    unsigned short* d = dst + (long)b * dstBS;
    const int tx = threadIdx.x, ty = threadIdx.y;   // 32 x 8
#pragma unroll
    for (int i = 0; i < 4; ++i)
        tile[ty + 8 * i][tx] = s[(long)(c0 + ty + 8 * i) * cols + n0 + tx];
    __syncthreads();
#pragma unroll
    for (int i = 0; i < 4; ++i)
        d[(long)(n0 + ty + 8 * i) * dstStride + dstColOff + c0 + tx] =
            f2h(tile[tx][ty + 8 * i]);
}

// ---------------- 2) elementwise f32->f16 (weights) -------------------------
__global__ void cvt_f16(const float* __restrict__ src, unsigned short* __restrict__ dst, int n) {
    int i = blockIdx.x * blockDim.x + threadIdx.x;
    if (i < n) dst[i] = f2h(src[i]);
}

// ---------------- 3) brute-force 3-NN over LDS-resident coarse cloud --------
__global__ void knn_top3(const float* __restrict__ xyz, const float* __restrict__ sub_xyz,
                         int* __restrict__ idx3, float* __restrict__ w3) {
    __shared__ float px[N_], py[N_], pz[N_];       // 48 KB of the 320 KB WGP LDS
    const int tid = blockIdx.x * blockDim.x + threadIdx.x;   // 0 .. B*M-1
    const int b = tid / M_;                        // uniform per block (M_ % 256 == 0)
    const int m = tid % M_;
    const float* sp = sub_xyz + (long)b * 3 * N_;
    for (int i = threadIdx.x; i < N_; i += blockDim.x) {
        px[i] = sp[i]; py[i] = sp[N_ + i]; pz[i] = sp[2 * N_ + i];
    }
    __syncthreads();

    const float* q = xyz + (long)b * 3 * M_;
    const float qx = q[m], qy = q[M_ + m], qz = q[2 * M_ + m];
    float d0 = 1e30f, d1 = 1e30f, d2 = 1e30f;
    int   i0 = 0, i1 = 0, i2 = 0;
#pragma unroll 4
    for (int n = 0; n < N_; ++n) {
        const float dx = qx - px[n], dy = qy - py[n], dz = qz - pz[n];
        const float d = fmaf(dx, dx, fmaf(dy, dy, dz * dz));
        if (d < d2) {
            if (d < d0)      { d2 = d1; i2 = i1; d1 = d0; i1 = i0; d0 = d; i0 = n; }
            else if (d < d1) { d2 = d1; i2 = i1; d1 = d;  i1 = n; }
            else             { d2 = d;  i2 = n; }
        }
    }
    float w0 = 1.0f / (d0 + 1e-8f);
    float w1 = 1.0f / (d1 + 1e-8f);
    float w2 = 1.0f / (d2 + 1e-8f);
    const float inv = 1.0f / (w0 + w1 + w2);
    idx3[tid * 3 + 0] = i0; idx3[tid * 3 + 1] = i1; idx3[tid * 3 + 2] = i2;
    w3[tid * 3 + 0] = w0 * inv; w3[tid * 3 + 1] = w1 * inv; w3[tid * 3 + 2] = w2 * inv;
}

// ---------------- 4) 3-point interpolation, one wave32 per query ------------
// subT: [B, N, CS_] f16 ; writes Y[b][m][128 .. 383] (f16)
__global__ void interp_write(const unsigned short* __restrict__ subT,
                             const int* __restrict__ idx3, const float* __restrict__ w3,
                             unsigned short* __restrict__ Y) {
    const int gq   = (blockIdx.x * blockDim.x + threadIdx.x) >> 5;  // query id
    const int lane = threadIdx.x & 31;
    const int b = gq / M_, m = gq % M_;
    const int i0 = idx3[gq * 3 + 0], i1 = idx3[gq * 3 + 1], i2 = idx3[gq * 3 + 2];
    const float w0 = w3[gq * 3 + 0], w1 = w3[gq * 3 + 1], w2 = w3[gq * 3 + 2];
    const unsigned short* base = subT + (long)b * N_ * CS_;
    const uint4 a = ((const uint4*)(base + (long)i0 * CS_))[lane];  // 8 f16 / lane
    const uint4 c = ((const uint4*)(base + (long)i1 * CS_))[lane];
    const uint4 e = ((const uint4*)(base + (long)i2 * CS_))[lane];

    uint4 o;
    o.x = packh2(w0 * hlo(a.x) + w1 * hlo(c.x) + w2 * hlo(e.x),
                 w0 * hhi(a.x) + w1 * hhi(c.x) + w2 * hhi(e.x));
    o.y = packh2(w0 * hlo(a.y) + w1 * hlo(c.y) + w2 * hlo(e.y),
                 w0 * hhi(a.y) + w1 * hhi(c.y) + w2 * hhi(e.y));
    o.z = packh2(w0 * hlo(a.z) + w1 * hlo(c.z) + w2 * hlo(e.z),
                 w0 * hhi(a.z) + w1 * hhi(c.z) + w2 * hhi(e.z));
    o.w = packh2(w0 * hlo(a.w) + w1 * hlo(c.w) + w2 * hlo(e.w),
                 w0 * hhi(a.w) + w1 * hhi(c.w) + w2 * hhi(e.w));
    ((uint4*)(Y + ((long)b * M_ + m) * K1_ + C_))[lane] = o;
}

// ---------------- 5) WMMA GEMM:  Z[b,o,m] = sum_k Wb[o,k] * Yb[b,m,k] -------
// Wb: [DO_, K] f16 row-major ; Yb: [B, M, K] f16 (K contiguous)
// 256 threads = 8 waves; wave w computes 16x16 tile at (o = 16w, m = 16*blk)
template <int K>
__global__ void gemm_wmma(const unsigned short* __restrict__ Wb,
                          const unsigned short* __restrict__ Yb,
                          float* __restrict__ Z) {
    const int lane = threadIdx.x & 31;
    const int wid  = threadIdx.x >> 5;
    const int mblk = blockIdx.x % (M_ / 16);
    const int b    = blockIdx.x / (M_ / 16);
    const int m0   = mblk * 16;
    const int o0   = wid * 16;
    const int h = lane >> 4;        // half-wave selector
    const int l = lane & 15;        // row (A) / column (B) within tile

    // A layout (16-bit 16x32): lane row = l, VGPR0-3: K = 8h+{0..7}, VGPR4-7: +16
    const unsigned short* arow = Wb + (long)(o0 + l) * K + h * 8;
    // B layout (16-bit 32x16): lane col = l, VGPR0-3: K = 8h+{0..7}, VGPR4-7: +16
    const unsigned short* brow = Yb + ((long)b * M_ + m0 + l) * K + h * 8;

    v8f acc = {0.f, 0.f, 0.f, 0.f, 0.f, 0.f, 0.f, 0.f};
#pragma unroll
    for (int kc = 0; kc < K; kc += 32) {
        if (kc + 32 < K) {
            __builtin_prefetch(brow + kc + 32, 0, 3);   // global_prefetch_b8, WGP scope
        }
        Frag32 fa, fb;
        fa.x = *(const uint4*)(arow + kc);        // K = kc + 8h + 0..7
        fa.y = *(const uint4*)(arow + kc + 16);   // K = kc + 8h + 16..23
        fb.x = *(const uint4*)(brow + kc);        // K = kc + 8h + 0..7
        fb.y = *(const uint4*)(brow + kc + 16);   // K = kc + 8h + 16..23
        const v16h A  = __builtin_bit_cast(v16h, fa);
        const v16h Bf = __builtin_bit_cast(v16h, fb);
        acc = __builtin_amdgcn_wmma_f32_16x16x32_f16(
            /*neg_a=*/false, A, /*neg_b=*/false, Bf,
            /*c_mod=*/(short)0, acc, /*reuse_a=*/false, /*reuse_b=*/false);
    }

    // C/D layout: VGPR r, lane -> (row = r + 8h, col = l)
    union { v8f v; float f[8]; } u; u.v = acc;
    float* zb = Z + ((long)b * DO_ + o0 + h * 8) * M_ + m0 + l;
#pragma unroll
    for (int r = 0; r < 8; ++r) zb[(long)r * M_] = u.f[r];
}

// ---------------- 6) BN batch stats -> fused scale/shift per channel --------
__global__ void bn_stats(const float* __restrict__ Z, const float* __restrict__ g,
                         const float* __restrict__ beta, float* __restrict__ scale,
                         float* __restrict__ shift) {
    const int o = blockIdx.x;                      // channel
    float s = 0.f, s2 = 0.f;
    for (int i = threadIdx.x; i < B_ * M_; i += blockDim.x) {
        const int b = i / M_, m = i % M_;
        const float z = Z[((long)b * DO_ + o) * M_ + m];
        s += z; s2 += z * z;
    }
    __shared__ float rs[256], rq[256];
    rs[threadIdx.x] = s; rq[threadIdx.x] = s2;
    __syncthreads();
    for (int st = 128; st > 0; st >>= 1) {
        if ((int)threadIdx.x < st) {
            rs[threadIdx.x] += rs[threadIdx.x + st];
            rq[threadIdx.x] += rq[threadIdx.x + st];
        }
        __syncthreads();
    }
    if (threadIdx.x == 0) {
        const float n    = (float)(B_ * M_);
        const float mean = rs[0] / n;
        const float var  = rq[0] / n - mean * mean;
        const float inv  = rsqrtf(var + 1e-5f);
        const float sc   = g[o] * inv;
        scale[o] = sc;
        shift[o] = beta[o] - mean * sc;
    }
}

// ---------------- 7) BN apply + ReLU + transpose to [b, m, o] f16 -----------
__global__ void bn_apply_T(const float* __restrict__ Z, const float* __restrict__ scale,
                           const float* __restrict__ shift, unsigned short* __restrict__ Y2) {
    __shared__ float tile[32][33];
    const int b  = blockIdx.z;
    const int o0 = blockIdx.y * 32;
    const int m0 = blockIdx.x * 32;
    const int tx = threadIdx.x, ty = threadIdx.y;  // 32 x 8
#pragma unroll
    for (int i = 0; i < 4; ++i) {
        const int o = o0 + ty + 8 * i;
        float z = Z[((long)b * DO_ + o) * M_ + m0 + tx];
        z = z * scale[o] + shift[o];
        tile[ty + 8 * i][tx] = z > 0.f ? z : 0.f;
    }
    __syncthreads();
#pragma unroll
    for (int i = 0; i < 4; ++i)
        Y2[((long)b * M_ + m0 + ty + 8 * i) * DO_ + o0 + tx] =
            f2h(tile[tx][ty + 8 * i]);
}

// ---------------- 8) final BN apply + ReLU in place (fp32 out) --------------
__global__ void bn_apply_out(float* __restrict__ Z, const float* __restrict__ scale,
                             const float* __restrict__ shift) {
    const int i = blockIdx.x * blockDim.x + threadIdx.x;   // < B*DO*M
    const int o = (i / M_) % DO_;
    const float z = Z[i] * scale[o] + shift[o];
    Z[i] = z > 0.f ? z : 0.f;
}

// ----------------------------------------------------------------------------
extern "C" void kernel_launch(void* const* d_in, const int* in_sizes, int n_in,
                              void* d_out, int out_size, void* d_ws, size_t ws_size,
                              hipStream_t stream) {
    const float* x       = (const float*)d_in[0];   // [B,C,M]
    const float* xyz     = (const float*)d_in[1];   // [B,3,M]
    const float* sub_x   = (const float*)d_in[2];   // [B,Cs,N]
    const float* sub_xyz = (const float*)d_in[3];   // [B,3,N]
    const float* W1      = (const float*)d_in[4];   // [Do, C+Cs]
    const float* g1      = (const float*)d_in[5];
    const float* b1      = (const float*)d_in[6];
    const float* W2      = (const float*)d_in[7];   // [Do, Do]
    const float* g2      = (const float*)d_in[8];
    const float* b2      = (const float*)d_in[9];
    float* out = (float*)d_out;                     // [B,Do,M] fp32; doubles as z2

    // ---- workspace partition (all chunks 256B-aligned sizes) ----
    char* p = (char*)d_ws;
    unsigned short* Y1   = (unsigned short*)p; p += (size_t)B_ * M_ * K1_ * 2;  // 25.2 MB
    unsigned short* subT = (unsigned short*)p; p += (size_t)B_ * N_ * CS_ * 2;  //  4.2 MB
    unsigned short* W1h  = (unsigned short*)p; p += (size_t)DO_ * K1_ * 2;      //  96 KB
    unsigned short* W2h  = (unsigned short*)p; p += (size_t)DO_ * DO_ * 2;      //  32 KB
    int*   idx3 = (int*)p;   p += (size_t)B_ * M_ * 3 * 4;                      // 384 KB
    float* w3   = (float*)p; p += (size_t)B_ * M_ * 3 * 4;                      // 384 KB
    float* z1   = (float*)p; p += (size_t)B_ * DO_ * M_ * 4;                    // 16.8 MB
    unsigned short* Y2 = (unsigned short*)p; p += (size_t)B_ * M_ * DO_ * 2;    //  8.4 MB
    float* sc1 = (float*)p; p += 256 * 4;
    float* sh1 = (float*)p; p += 256 * 4;
    float* sc2 = (float*)p; p += 256 * 4;
    float* sh2 = (float*)p; p += 256 * 4;

    const dim3 tb(32, 8);

    // 1) x [B,C,M] -> Y1[b][m][0..127] f16
    transpose_cvt<<<dim3(M_ / 32, C_ / 32, B_), tb, 0, stream>>>(
        x, Y1, M_, (long)C_ * M_, (long)M_ * K1_, K1_, 0);
    // 2) sub_x [B,Cs,N] -> subT [B,N,Cs] f16
    transpose_cvt<<<dim3(N_ / 32, CS_ / 32, B_), tb, 0, stream>>>(
        sub_x, subT, N_, (long)CS_ * N_, (long)N_ * CS_, CS_, 0);
    // 3) weights -> f16
    cvt_f16<<<(DO_ * K1_ + 255) / 256, 256, 0, stream>>>(W1, W1h, DO_ * K1_);
    cvt_f16<<<(DO_ * DO_ + 255) / 256, 256, 0, stream>>>(W2, W2h, DO_ * DO_);
    // 4) 3-NN (LDS-resident coarse cloud)
    knn_top3<<<(B_ * M_) / 256, 256, 0, stream>>>(xyz, sub_xyz, idx3, w3);
    // 5) interpolate -> Y1[b][m][128..383]
    interp_write<<<(B_ * M_) / 8, 256, 0, stream>>>(subT, idx3, w3, Y1);
    // 6) GEMM1 (K=384) -> z1 [B,Do,M]
    gemm_wmma<K1_><<<B_ * (M_ / 16), 256, 0, stream>>>(W1h, Y1, z1);
    // 7) BN1 stats + apply(+ReLU) with transpose -> Y2 [B,M,Do] f16
    bn_stats<<<DO_, 256, 0, stream>>>(z1, g1, b1, sc1, sh1);
    bn_apply_T<<<dim3(M_ / 32, DO_ / 32, B_), tb, 0, stream>>>(z1, sc1, sh1, Y2);
    // 8) GEMM2 (K=128) -> z2 stored directly in d_out
    gemm_wmma<DO_><<<B_ * (M_ / 16), 256, 0, stream>>>(W2h, Y2, out);
    // 9) BN2 stats + apply(+ReLU) in place -> final fp32 output [B,Do,M]
    bn_stats<<<DO_, 256, 0, stream>>>(out, g2, b2, sc2, sh2);
    bn_apply_out<<<(B_ * DO_ * M_) / 256, 256, 0, stream>>>(out, sc2, sh2);
}
